// RelationalGCNLayer_28724741275659
// MI455X (gfx1250) — matrix operand order repaired and checked by reading, not compiled, not run on hardware
//
#include <hip/hip_runtime.h>

// Problem constants (from reference): B,R,N,D_IN,D_OUT = 8,8,1024,256,256
#define BB   8
#define RR   8
#define NN   1024
#define DIN  256
#define DOUT 256

typedef __attribute__((ext_vector_type(16))) __bf16 v16bf;
typedef __attribute__((ext_vector_type(8)))  __bf16 v8bf;
typedef __attribute__((ext_vector_type(8)))  float  v8f;
typedef __attribute__((ext_vector_type(4)))  float  v4f;

// ---------------------------------------------------------------------------
// Kernel 0: W [R, DIN, DOUT] f32  ->  WT [R, DOUT, DIN] bf16  (B-operand prep)
// ---------------------------------------------------------------------------
__global__ __launch_bounds__(256) void rgcn_wt_kernel(
    const float* __restrict__ Wsrc, __bf16* __restrict__ WT) {
  int idx = blockIdx.x * 256 + threadIdx.x;          // over R*DOUT*DIN, write-coalesced
  if (idx >= RR * DIN * DOUT) return;
  int d = idx & (DIN - 1);
  int e = (idx >> 8) & (DOUT - 1);
  int r = idx >> 16;
  WT[idx] = (__bf16)Wsrc[((size_t)r * DIN + d) * DOUT + e];
}

// Load 8 consecutive f32 (global or LDS) -> bf16 fragment slots [base..base+7]
__device__ __forceinline__ void load8_cvt(const float* p, v16bf& frag, int base) {
  v4f lo = *(const v4f*)p;
  v4f hi = *(const v4f*)(p + 4);
#pragma unroll
  for (int i = 0; i < 4; ++i) {
    frag[base + i]     = (__bf16)lo[i];
    frag[base + 4 + i] = (__bf16)hi[i];
  }
}

// One async 16B-per-lane copy: global -> LDS (tracked by ASYNCcnt)
__device__ __forceinline__ void async_tile_load(unsigned int lds_off,
                                                const float* gptr) {
  unsigned long long ga = (unsigned long long)gptr;
  asm volatile("global_load_async_to_lds_b128 %0, %1, off"
               :: "v"(lds_off), "v"(ga) : "memory");
}

// ---------------------------------------------------------------------------
// Kernel 1: supportT[b,r,e,n] = bf16( x[b] @ W[r] + bias[r] )
//   block = (b, r, n-tile of 16); 4 waves, wave w owns e in [w*64, w*64+64)
// ---------------------------------------------------------------------------
__global__ __launch_bounds__(128) void rgcn_support_kernel(
    const float*  __restrict__ x,        // [B, N, DIN]
    const __bf16* __restrict__ WT,       // [R, DOUT, DIN]
    const float*  __restrict__ bias,     // [R, DOUT]
    __bf16*       __restrict__ supportT) // [B, R, DOUT, N]
{
  const int lane = threadIdx.x & 31;
  const int wave = threadIdx.x >> 5;
  const int half = lane >> 4;            // 0 or 1
  const int l15  = lane & 15;

  const int blk   = blockIdx.x;          // B*R*64 blocks
  const int ntile = blk & 63;
  const int r     = (blk >> 6) & 7;
  const int b     = blk >> 9;
  const int n0    = ntile * 16;
  const int e0    = wave * 64;

  const float* xrow = x + ((size_t)b * NN + (n0 + l15)) * DIN;

  v8f acc[4] = {};

#pragma unroll 2
  for (int k0 = 0; k0 < DIN; k0 += 32) {
    v16bf a;
    load8_cvt(xrow + k0 + 16 * half, a, 0);
    load8_cvt(xrow + k0 + 16 * half + 8, a, 8);
#pragma unroll
    for (int et = 0; et < 4; ++et) {
      const __bf16* pb = WT + ((size_t)r * DOUT + (e0 + et * 16 + l15)) * DIN
                            + k0 + 16 * half;
      v16bf bm = *(const v16bf*)pb;
      acc[et] = __builtin_amdgcn_wmma_f32_16x16x32_bf16(
          false, a, false, bm, (short)0, acc[et], false, false);
    }
  }

#pragma unroll
  for (int et = 0; et < 4; ++et) {
    const int e = e0 + et * 16 + l15;
    const float bb = bias[r * DOUT + e];
    v8bf outv;
#pragma unroll
    for (int v = 0; v < 8; ++v) outv[v] = (__bf16)(acc[et][v] + bb);
    __bf16* po = supportT + (((size_t)b * RR + r) * DOUT + e) * NN + n0 + 8 * half;
    *(v8bf*)po = outv;
  }
}

// ---------------------------------------------------------------------------
// Kernel 2: out[b,m,e] = relu( sum_r adj[b,r,m,:] @ support[b,r,:,e] )
//   block = (b, m-tile of 16); 4 waves, wave w owns e in [w*64, w*64+64)
//   adj tile staged via async-to-LDS ring (4 x 2KB); B fragments register
//   double-buffered one step ahead. 256 pipelined K-steps (r x n0).
// ---------------------------------------------------------------------------
__global__ __launch_bounds__(128) void rgcn_aggregate_kernel(
    const float*  __restrict__ adj,      // [B, R, N, N]
    const __bf16* __restrict__ supportT, // [B, R, DOUT, N]
    float*        __restrict__ out)      // [B, N, DOUT]
{
  __shared__ float lds_adj[4][16 * 32];  // 4-deep ring of 16x32 f32 tiles

  const int t    = threadIdx.x;          // 0..127
  const int lane = t & 31;
  const int wave = t >> 5;
  const int half = lane >> 4;
  const int l15  = lane & 15;

  const int blk   = blockIdx.x;          // B*64 blocks
  const int mtile = blk & 63;
  const int b     = blk >> 6;
  const int m0    = mtile * 16;
  const int e0    = wave * 64;

  // Async source: lane t copies 16B = row (t>>3), cols (t&7)*4 .. +3 of the tile.
  const int arow = t >> 3;
  const int acol = (t & 7) * 4;
  const float* gsrc = adj + (((size_t)b * RR) * NN + (m0 + arow)) * NN + acol;

  // Per-lane base into supportT for B fragments (add r*DOUT*NN + et*16*NN + n0).
  const __bf16* sb = supportT + (size_t)b * RR * DOUT * NN
                   + (size_t)(e0 + l15) * NN + 16 * half;

  v16bf breg[2][4];
  v8f acc[4] = {};

  // ---- prologue: stage tile 0 (r=0, n0=0) ----
  async_tile_load((unsigned int)(unsigned long long)&lds_adj[0][t * 4], gsrc);
#pragma unroll
  for (int et = 0; et < 4; ++et)
    breg[0][et] = *(const v16bf*)(sb + (size_t)et * 16 * NN);

  // ---- 256 pipelined steps: s -> (r = s>>5, n0 = (s&31)*32) ----
  for (int s0 = 0; s0 < 256; s0 += 4) {
#pragma unroll
    for (int u = 0; u < 4; ++u) {
      const int s   = s0 + u;
      const int nxt = (s + 1) & 255;     // wraparound prefetch: branch-free body
      const int rn  = nxt >> 5;
      const int nn0 = (nxt & 31) << 5;

      // prefetch A tile s+1 -> LDS ring slot (u+1)&3 (one async instr per wave)
      async_tile_load(
          (unsigned int)(unsigned long long)&lds_adj[(u + 1) & 3][t * 4],
          gsrc + ((size_t)rn << 20) + nn0);

      // prefetch B fragments for step s+1 into the other register bank
      {
        const __bf16* pb = sb + (size_t)rn * DOUT * NN + nn0;
#pragma unroll
        for (int et = 0; et < 4; ++et)
          breg[(u + 1) & 1][et] = *(const v16bf*)(pb + (size_t)et * 16 * NN);
      }

      // tile s has landed once our own count drops to 1; barrier covers all waves
      asm volatile("s_wait_asynccnt 0x1" ::: "memory");
      __syncthreads();

      // A fragment from LDS slot u&3 (row l15, K 16h..16h+15)
      v16bf a;
      const float* lrow = &lds_adj[u & 3][l15 * 32 + 16 * half];
      load8_cvt(lrow, a, 0);
      load8_cvt(lrow + 8, a, 8);

#pragma unroll
      for (int et = 0; et < 4; ++et)
        acc[et] = __builtin_amdgcn_wmma_f32_16x16x32_bf16(
            false, a, false, breg[u & 1][et], (short)0, acc[et], false, false);
    }
  }

  // ---- ReLU + store: lane holds column e, rows m = m0 + 8*half + v ----
#pragma unroll
  for (int et = 0; et < 4; ++et) {
    const int e = e0 + et * 16 + l15;
#pragma unroll
    for (int v = 0; v < 8; ++v) {
      float res = acc[et][v];
      res = res > 0.0f ? res : 0.0f;
      out[((size_t)b * NN + (m0 + 8 * half + v)) * DOUT + e] = res;
    }
  }
}

// ---------------------------------------------------------------------------
// Launch: d_in = { x, adj_stack, W, b } (setup_inputs order), all f32.
// Workspace: WT (1 MB) | supportT (33.5 MB).
// ---------------------------------------------------------------------------
extern "C" void kernel_launch(void* const* d_in, const int* in_sizes, int n_in,
                              void* d_out, int out_size, void* d_ws, size_t ws_size,
                              hipStream_t stream) {
  (void)in_sizes; (void)n_in; (void)out_size; (void)ws_size;

  const float* x    = (const float*)d_in[0];
  const float* adj  = (const float*)d_in[1];
  const float* W    = (const float*)d_in[2];
  const float* bias = (const float*)d_in[3];
  float* out = (float*)d_out;

  __bf16* WT       = (__bf16*)d_ws;
  __bf16* supportT = (__bf16*)((char*)d_ws + (size_t)RR * DOUT * DIN * sizeof(__bf16));

  rgcn_wt_kernel<<<(RR * DIN * DOUT + 255) / 256, 256, 0, stream>>>(W, WT);
  rgcn_support_kernel<<<BB * RR * (NN / 16), 128, 0, stream>>>(x, WT, bias, supportT);
  rgcn_aggregate_kernel<<<BB * (NN / 16), 128, 0, stream>>>(adj, supportT, out);
}
